// DGMatch_38568806318768
// MI455X (gfx1250) — compile-verified, matrix operand order checked by reading, not run on hardware
//
#include <hip/hip_runtime.h>
#include <hip/hip_bf16.h>

// ---------------------------------------------------------------------------
// Problem constants (match the reference)
// ---------------------------------------------------------------------------
#define Bb 16
#define Nn 2048
#define Ee 64
#define Kk 16
#define Ll 3
#define Cc 21
#define PPW 4   // points per wave in edgeconv (amortizes weight-fragment loads)

typedef _Float16 half_t;
typedef __attribute__((ext_vector_type(16))) _Float16 v16h;
typedef __attribute__((ext_vector_type(8)))  float    v8f;

// ---------------------------------------------------------------------------
// Workspace layout (bytes). Total ~34.5 MB.
// ---------------------------------------------------------------------------
static constexpr size_t SZH     = (size_t)Bb * Nn * 128 * sizeof(half_t);    // 8 MB per layer-rep
static constexpr size_t SQ_OFF  = 4 * SZH;
static constexpr size_t IDX_OFF = SQ_OFF  + (size_t)Bb * Nn * sizeof(float);
static constexpr size_t PG_OFF  = IDX_OFF + (size_t)Bb * Nn * Kk * sizeof(int);
static constexpr size_t PG_SZ   = (size_t)256 * 128 * sizeof(half_t);        // 64 KB each
static constexpr size_t PF_OFF  = PG_OFF  + 3 * PG_SZ;
static constexpr size_t PF_SZ   = (size_t)448 * 128 * sizeof(half_t);
static constexpr size_t PR0_OFF = PF_OFF  + PF_SZ;                           // 128x64
static constexpr size_t PR1_OFF = PR0_OFF + (size_t)128 * 64 * 2;            // 64x64
static constexpr size_t PR2_OFF = PR1_OFF + (size_t)64 * 64 * 2;             // 64x32
static constexpr size_t PC0_OFF = PR2_OFF + (size_t)64 * 32 * 2;             // 128x64
static constexpr size_t PC1_OFF = PC0_OFF + (size_t)128 * 64 * 2;            // 64x16
static constexpr size_t PC2_OFF = PC1_OFF + (size_t)64 * 16 * 2;             // 32x32 (padded 16x21)
static constexpr size_t PO0_OFF = PC2_OFF + (size_t)32 * 32 * 2;             // 128x64

// ---------------------------------------------------------------------------
// WMMA helpers (CDNA5 wave32, D = A(16x32 f16) * B(32x16 f16) + C(16x16 f32))
// ---------------------------------------------------------------------------
__device__ __forceinline__ v8f wmma16x16x32(v16h a, v16h b, v8f c) {
  return __builtin_amdgcn_wmma_f32_16x16x32_f16(
      /*neg_a=*/false, a, /*neg_b=*/false, b,
      /*c_mod=*/(short)0, c, /*reuse_a=*/false, /*reuse_b=*/false);
}

// A-fragment (16x32, 16-bit) from a row-major matrix X[row][k], leading dim ld.
// ISA 7.12.2: lanes 0-15 carry M=0..15 with K=k0..k0+7 / k0+16..k0+23,
// lanes 16-31 the same M with K shifted by +8.  For B = X^T (Gram) the
// B-fragment gather is the identical pattern with "row" = column index.
__device__ __forceinline__ v16h load_frag_rm(const half_t* X, int ld, int r0, int k0) {
  const int lane = threadIdx.x & 31;
  const int m    = lane & 15;
  const int koff = (lane & 16) ? 8 : 0;
  const half_t* p = X + (size_t)(r0 + m) * ld + (k0 + koff);
  v16h v;
#pragma unroll
  for (int i = 0; i < 8; ++i) { v[i] = p[i]; v[8 + i] = p[16 + i]; }
  return v;
}

// B-fragment from pre-packed weights: fragment (ks,nt) for lane L is 16
// contiguous halves (32 B) -> one v16h load, no strided gather.
__device__ __forceinline__ v16h load_fragB_packed(const half_t* __restrict__ P,
                                                  int nn, int ks, int nt) {
  const int lane = threadIdx.x & 31;
  return *(const v16h*)(P + ((size_t)(ks * nn + nt) * 32 + lane) * 16);
}

// ---------------------------------------------------------------------------
// Pack a row-major f32 weight W[K][N] into WMMA-B fragment order (f16),
// zero-padding to (Kp,Np).  One thread per (fragment, lane).
// ---------------------------------------------------------------------------
__global__ void pack_b_kernel(const float* __restrict__ W, half_t* __restrict__ P,
                              int K, int N, int Kp, int Np) {
  const int total = (Kp >> 5) * (Np >> 4) * 32;
  const int tid = blockIdx.x * blockDim.x + threadIdx.x;
  if (tid >= total) return;
  const int lane = tid & 31;
  const int f    = tid >> 5;
  const int nn   = Np >> 4;
  const int ks   = f / nn;
  const int nt   = f - ks * nn;
  const int n    = nt * 16 + (lane & 15);
  const int koff = (lane & 16) ? 8 : 0;
  half_t* dst = P + ((size_t)f * 32 + lane) * 16;
#pragma unroll
  for (int i = 0; i < 8; ++i) {
    const int k1 = ks * 32 + koff + i;
    const int k2 = ks * 32 + 16 + koff + i;
    dst[i]     = (k1 < K && n < N) ? (half_t)W[(size_t)k1 * N + n] : (half_t)0.f;
    dst[8 + i] = (k2 < K && n < N) ? (half_t)W[(size_t)k2 * N + n] : (half_t)0.f;
  }
}

// Generic GEMM layer: A (16xK f16, LDS) x packed B -> ReLU -> f16 LDS dst.
template <int KS, int NT>
__device__ __forceinline__ void gemm_relu_layer(const half_t* A, int ldA,
                                                const half_t* __restrict__ P,
                                                const float* __restrict__ bias,
                                                half_t* dst, int ldD,
                                                int mbase, int nlo) {
  for (int nt = 0; nt < NT; ++nt) {
    v8f acc = {};
    for (int ks = 0; ks < KS; ++ks)
      acc = wmma16x16x32(load_frag_rm(A, ldA, 0, ks * 32),
                         load_fragB_packed(P, NT, ks, nt), acc);
    const int   n  = nt * 16 + nlo;
    const float bn = bias[n];
#pragma unroll
    for (int r = 0; r < 8; ++r)
      dst[(size_t)(mbase + r) * ldD + n] = (half_t)fmaxf(acc[r] + bn, 0.f);
  }
  __syncthreads();
}

// ---------------------------------------------------------------------------
// Embedding: class_emb[ci] + pos_mlp(pos), concat desc -> H0 (f16) + |x|^2
// ---------------------------------------------------------------------------
__global__ void embed_kernel(const int* __restrict__ cls, const float* __restrict__ pos,
                             const float* __restrict__ desc, const float* __restrict__ cemb,
                             const float* W0, const float* b0, const float* W1, const float* b1,
                             const float* W2, const float* b2, const float* W3, const float* b3,
                             half_t* __restrict__ H0, float* __restrict__ SQ) {
  const int p = blockIdx.x * blockDim.x + threadIdx.x;
  if (p >= Bb * Nn) return;
  const int b = p / Nn;
  const float x0 = pos[2 * p + 0], x1 = pos[2 * p + 1];
  float h4[4], h8[8], h16[16];
#pragma unroll
  for (int j = 0; j < 4; ++j)
    h4[j] = fmaxf(b0[j] + x0 * W0[0 * 4 + j] + x1 * W0[1 * 4 + j], 0.f);
#pragma unroll
  for (int j = 0; j < 8; ++j) {
    float s = b1[j];
#pragma unroll
    for (int i = 0; i < 4; ++i) s += h4[i] * W1[i * 8 + j];
    h8[j] = fmaxf(s, 0.f);
  }
#pragma unroll
  for (int j = 0; j < 16; ++j) {
    float s = b2[j];
#pragma unroll
    for (int i = 0; i < 8; ++i) s += h8[i] * W2[i * 16 + j];
    h16[j] = fmaxf(s, 0.f);
  }
  const int ci = cls[p];
  float sq = 0.f;
  for (int j = 0; j < 64; ++j) {
    float s = b3[j] + cemb[ci * 64 + j];
    for (int i = 0; i < 16; ++i) s += h16[i] * W3[i * 64 + j];
    half_t hv = (half_t)s;
    H0[(size_t)p * 128 + j] = hv;
    float vf = (float)hv; sq += vf * vf;
  }
  for (int j = 0; j < 64; ++j) {
    half_t hv = (half_t)desc[b * 64 + j];
    H0[(size_t)p * 128 + 64 + j] = hv;
    float vf = (float)hv; sq += vf * vf;
  }
  SQ[p] = sq;
}

// ---------------------------------------------------------------------------
// kNN: one wave per 16 rows of one batch.  Gram tile via 4 chained WMMAs
// (K=128), d2 in LDS, per-lane sorted top-16 insertion over all columns.
// ---------------------------------------------------------------------------
__global__ __launch_bounds__(32) void knn_kernel(const half_t* __restrict__ H,
                                                 const float* __restrict__ SQ,
                                                 int* __restrict__ IDX) {
  __shared__ float tile[16][17];
  const int lane = threadIdx.x;
  const int nt   = Nn / 16;
  const int b    = blockIdx.x / nt;
  const int r0   = (blockIdx.x % nt) * 16;
  const half_t* Xb  = H  + (size_t)b * Nn * 128;
  const float*  sqb = SQ + (size_t)b * Nn;

  const v16h a0 = load_frag_rm(Xb, 128, r0, 0);
  const v16h a1 = load_frag_rm(Xb, 128, r0, 32);
  const v16h a2 = load_frag_rm(Xb, 128, r0, 64);
  const v16h a3 = load_frag_rm(Xb, 128, r0, 96);

  const int mbase = (lane >> 4) * 8;
  float sqr[8];
#pragma unroll
  for (int r = 0; r < 8; ++r) sqr[r] = sqb[r0 + mbase + r];

  float bd[Kk]; int bi[Kk];
#pragma unroll
  for (int t = 0; t < Kk; ++t) { bd[t] = 3.0e38f; bi[t] = 0; }

  for (int ct = 0; ct < nt; ++ct) {
    const int c0 = ct * 16;
    if (ct + 1 < nt)
      __builtin_prefetch(Xb + (size_t)(c0 + 16) * 128 + (lane << 4), 0, 3);

    v8f acc = {};
    acc = wmma16x16x32(a0, load_frag_rm(Xb, 128, c0, 0),  acc);
    acc = wmma16x16x32(a1, load_frag_rm(Xb, 128, c0, 32), acc);
    acc = wmma16x16x32(a2, load_frag_rm(Xb, 128, c0, 64), acc);
    acc = wmma16x16x32(a3, load_frag_rm(Xb, 128, c0, 96), acc);

    const float sqc = sqb[c0 + (lane & 15)];
#pragma unroll
    for (int r = 0; r < 8; ++r)
      tile[mbase + r][lane & 15] = sqr[r] + sqc - 2.0f * acc[r];
    __syncthreads();

    if (lane < 16) {
      for (int j = 0; j < 16; ++j) {
        const float d  = tile[lane][j];
        const int   ci = c0 + j;
        if (d < bd[Kk - 1]) {
          bd[Kk - 1] = d; bi[Kk - 1] = ci;
#pragma unroll
          for (int t = Kk - 1; t > 0; --t) {
            if (bd[t] < bd[t - 1]) {
              float td = bd[t]; bd[t] = bd[t - 1]; bd[t - 1] = td;
              int   ti = bi[t]; bi[t] = bi[t - 1]; bi[t - 1] = ti;
            }
          }
        }
      }
    }
    __syncthreads();
  }

  if (lane < 16) {
    int* op = IDX + ((size_t)b * Nn + r0 + lane) * Kk;
#pragma unroll
    for (int t = 0; t < Kk; ++t) op[t] = bi[t];
  }
}

// ---------------------------------------------------------------------------
// EdgeConv: one wave per PPW points.  msg (PPW x 16 nbrs x 256) in LDS.
// Per n-tile the 8 weight fragments are hoisted into VGPRs once and reused
// across the PPW points (4x less L2 weight traffic); max over neighbors via
// accumulator VGPRs + shfl_xor(16).
// ---------------------------------------------------------------------------
__global__ __launch_bounds__(32) void edgeconv_kernel(const half_t* __restrict__ Hin,
                                                      const int* __restrict__ IDX,
                                                      const half_t* __restrict__ PW,
                                                      const float* __restrict__ bias,
                                                      half_t* __restrict__ Hout,
                                                      float* __restrict__ SQ) {
  __shared__ half_t msg[PPW][16][256];
  const int lane = threadIdx.x;
  const int p0   = blockIdx.x * PPW;
  const int b    = p0 / Nn;                 // PPW divides Nn: same batch
  const int c0   = (lane & 15) * 8;

  for (int pt = 0; pt < PPW; ++pt) {
    const int p = p0 + pt;
    const half_t* xi = Hin + (size_t)p * 128;
    for (int m = 0; m < Kk; ++m) {
      const int j = IDX[(size_t)p * Kk + m];
      const half_t* xj = Hin + ((size_t)b * Nn + j) * 128;
      if (lane < 16) {
#pragma unroll
        for (int i = 0; i < 8; ++i) msg[pt][m][c0 + i] = xi[c0 + i];
      } else {
#pragma unroll
        for (int i = 0; i < 8; ++i)
          msg[pt][m][128 + c0 + i] = (half_t)((float)xj[c0 + i] - (float)xi[c0 + i]);
      }
    }
  }
  __syncthreads();

  float s[PPW];
#pragma unroll
  for (int pt = 0; pt < PPW; ++pt) s[pt] = 0.f;

  for (int ntile = 0; ntile < 8; ++ntile) {
    // Hoist this n-tile's weight fragments; reuse across PPW points.
    v16h bf[8];
#pragma unroll
    for (int ks = 0; ks < 8; ++ks) bf[ks] = load_fragB_packed(PW, 8, ks, ntile);

    const int   n  = ntile * 16 + (lane & 15);
    const float bn = bias[n];

    for (int pt = 0; pt < PPW; ++pt) {
      v8f acc = {};
#pragma unroll
      for (int ks = 0; ks < 8; ++ks)
        acc = wmma16x16x32(load_frag_rm(&msg[pt][0][0], 256, 0, ks * 32), bf[ks], acc);
      float vmax = acc[0] + bn;
#pragma unroll
      for (int r = 1; r < 8; ++r) vmax = fmaxf(vmax, acc[r] + bn);
      vmax = fmaxf(vmax, __shfl_xor(vmax, 16, 32));  // combine M=0..7 with M=8..15
      if (lane < 16) {
        half_t hv = (half_t)vmax;
        Hout[(size_t)(p0 + pt) * 128 + n] = hv;
        float vf = (float)hv; s[pt] += vf * vf;
      }
    }
  }

#pragma unroll
  for (int pt = 0; pt < PPW; ++pt) {
    float v = s[pt];
#pragma unroll
    for (int off = 1; off < 32; off <<= 1) v += __shfl_xor(v, off, 32);
    if (lane == 0) SQ[p0 + pt] = v;
  }
}

// ---------------------------------------------------------------------------
// feat_mlp: one wave per 16 points; A = [H1|H2|H3|desc] (16x448) in LDS,
// 8 n-tiles x 14 k-steps of WMMA, f32 output straight to d_out.
// ---------------------------------------------------------------------------
__global__ __launch_bounds__(32) void featmlp_kernel(const half_t* __restrict__ H1,
                                                     const half_t* __restrict__ H2,
                                                     const half_t* __restrict__ H3,
                                                     const float* __restrict__ desc,
                                                     const half_t* __restrict__ PW,
                                                     const float* __restrict__ bias,
                                                     float* __restrict__ out) {
  __shared__ half_t A[16][448];
  const int lane = threadIdx.x;
  const int p0   = blockIdx.x * 16;

  for (int m = 0; m < 16; ++m) {
    const int p = p0 + m;
    const int b = p / Nn;
    for (int c = lane; c < 448; c += 32) {
      half_t v;
      if      (c < 128) v = H1[(size_t)p * 128 + c];
      else if (c < 256) v = H2[(size_t)p * 128 + (c - 128)];
      else if (c < 384) v = H3[(size_t)p * 128 + (c - 256)];
      else              v = (half_t)desc[b * 64 + (c - 384)];
      A[m][c] = v;
    }
  }
  __syncthreads();

  const int mrow = (lane >> 4) * 8;
  for (int ntile = 0; ntile < 8; ++ntile) {
    v8f acc = {};
    for (int ks = 0; ks < 14; ++ks)
      acc = wmma16x16x32(load_frag_rm(&A[0][0], 448, 0, ks * 32),
                         load_fragB_packed(PW, 8, ks, ntile), acc);
    const int   n  = ntile * 16 + (lane & 15);
    const float bn = bias[n];
#pragma unroll
    for (int r = 0; r < 8; ++r)
      out[(size_t)(p0 + mrow + r) * 128 + n] = acc[r] + bn;
  }
}

// ---------------------------------------------------------------------------
// Heads via WMMA: one wave per 16 points, layers chained through LDS.
// ref: 128->64->64->32->1, objcls: 128->64->16->21(pad 32x32), off: 128->64->2
// ---------------------------------------------------------------------------
__global__ __launch_bounds__(32) void heads_kernel(const float* __restrict__ F,
    const half_t* PR0, const float* rb0, const half_t* PR1, const float* rb1,
    const half_t* PR2, const float* rb2, const float* rW3, const float* rb3,
    const half_t* PC0, const float* cb0, const half_t* PC1, const float* cb1,
    const half_t* PC2, const float* cb2,
    const half_t* PO0, const float* ob0, const float* oW1, const float* ob1,
    float* __restrict__ out_ref, float* __restrict__ out_cls,
    float* __restrict__ out_off) {
  __shared__ half_t Af[16][128];
  __shared__ half_t Ha[16][64];
  __shared__ half_t Hb[16][64];
  const int lane  = threadIdx.x;
  const int p0    = blockIdx.x * 16;
  const int mbase = (lane >> 4) * 8;
  const int nlo   = lane & 15;

  for (int m = 0; m < 16; ++m)
    for (int c = lane; c < 128; c += 32)
      Af[m][c] = (half_t)F[(size_t)(p0 + m) * 128 + c];
  __syncthreads();

  // ---- ref head: 128 -> 64 -> 64 -> 32 -> 1
  gemm_relu_layer<4, 4>(&Af[0][0], 128, PR0, rb0, &Ha[0][0], 64, mbase, nlo);
  gemm_relu_layer<2, 4>(&Ha[0][0], 64,  PR1, rb1, &Hb[0][0], 64, mbase, nlo);
  gemm_relu_layer<2, 2>(&Hb[0][0], 64,  PR2, rb2, &Ha[0][0], 64, mbase, nlo); // cols 0..31
  if (lane < 16) {
    float s = rb3[0];
    for (int i = 0; i < 32; ++i) s += (float)Ha[lane][i] * rW3[i];
    out_ref[p0 + lane] = s;
  }
  __syncthreads();

  // ---- objcls head: 128 -> 64 -> 16 -> 21
  gemm_relu_layer<4, 4>(&Af[0][0], 128, PC0, cb0, &Hb[0][0], 64, mbase, nlo);
  gemm_relu_layer<2, 1>(&Hb[0][0], 64,  PC1, cb1, &Ha[0][0], 64, mbase, nlo);
  // cols 16..31 of Ha hold stale (finite) ref values; PC2 zero-pads K>=16.
  for (int nt = 0; nt < 2; ++nt) {
    v8f acc = {};
    acc = wmma16x16x32(load_frag_rm(&Ha[0][0], 64, 0, 0),
                       load_fragB_packed(PC2, 2, 0, nt), acc);
    const int n = nt * 16 + nlo;
    if (n < Cc) {
      const float bn = cb2[n];
#pragma unroll
      for (int r = 0; r < 8; ++r)
        out_cls[(size_t)(p0 + mbase + r) * Cc + n] = acc[r] + bn;
    }
  }
  __syncthreads();

  // ---- off head: 128 -> 64 -> 2
  gemm_relu_layer<4, 4>(&Af[0][0], 128, PO0, ob0, &Hb[0][0], 64, mbase, nlo);
  if (lane < 16) {
#pragma unroll
    for (int j = 0; j < 2; ++j) {
      float s = ob1[j];
      for (int i = 0; i < 64; ++i) s += (float)Hb[lane][i] * oW1[i * 2 + j];
      out_off[(size_t)(p0 + lane) * 2 + j] = s;
    }
  }
}

// ---------------------------------------------------------------------------
// tgt_mlp on desc: 64 -> 16 -> 21 (one thread per batch)
// ---------------------------------------------------------------------------
__global__ void tgt_kernel(const float* __restrict__ desc,
                           const float* W0, const float* b0,
                           const float* W1, const float* b1,
                           float* __restrict__ out) {
  const int b = threadIdx.x;
  if (b >= Bb) return;
  float h[16];
  for (int j = 0; j < 16; ++j) {
    float s = b0[j];
    for (int i = 0; i < 64; ++i) s += desc[b * 64 + i] * W0[i * 16 + j];
    h[j] = fmaxf(s, 0.f);
  }
  for (int j = 0; j < Cc; ++j) {
    float s = b1[j];
    for (int i = 0; i < 16; ++i) s += h[i] * W1[i * Cc + j];
    out[b * Cc + j] = s;
  }
}

// ---------------------------------------------------------------------------
// Launch
// ---------------------------------------------------------------------------
extern "C" void kernel_launch(void* const* d_in, const int* in_sizes, int n_in,
                              void* d_out, int out_size, void* d_ws, size_t ws_size,
                              hipStream_t stream) {
  (void)in_sizes; (void)n_in; (void)out_size; (void)ws_size;

  const int*   cls  = (const int*)  d_in[0];
  const float* pos  = (const float*)d_in[1];
  const float* desc = (const float*)d_in[2];
  /* d_in[3] = k (fixed 16) */
  const float* cemb = (const float*)d_in[4];
  const float* pW0 = (const float*)d_in[5],  *pb0 = (const float*)d_in[6];
  const float* pW1 = (const float*)d_in[7],  *pb1 = (const float*)d_in[8];
  const float* pW2 = (const float*)d_in[9],  *pb2 = (const float*)d_in[10];
  const float* pW3 = (const float*)d_in[11], *pb3 = (const float*)d_in[12];
  const float* gW[3] = {(const float*)d_in[13], (const float*)d_in[15], (const float*)d_in[17]};
  const float* gB[3] = {(const float*)d_in[14], (const float*)d_in[16], (const float*)d_in[18]};
  const float* fW  = (const float*)d_in[19], *fB  = (const float*)d_in[20];
  const float* rW0 = (const float*)d_in[21], *rb0 = (const float*)d_in[22];
  const float* rW1 = (const float*)d_in[23], *rb1 = (const float*)d_in[24];
  const float* rW2 = (const float*)d_in[25], *rb2 = (const float*)d_in[26];
  const float* rW3 = (const float*)d_in[27], *rb3 = (const float*)d_in[28];
  const float* tW0 = (const float*)d_in[29], *tb0 = (const float*)d_in[30];
  const float* tW1 = (const float*)d_in[31], *tb1 = (const float*)d_in[32];
  const float* cW0 = (const float*)d_in[33], *cb0 = (const float*)d_in[34];
  const float* cW1 = (const float*)d_in[35], *cb1 = (const float*)d_in[36];
  const float* cW2 = (const float*)d_in[37], *cb2 = (const float*)d_in[38];
  const float* oW0 = (const float*)d_in[39], *ob0 = (const float*)d_in[40];
  const float* oW1 = (const float*)d_in[41], *ob1 = (const float*)d_in[42];

  float* out_feats = (float*)d_out;
  float* out_ref   = out_feats + (size_t)Bb * Nn * 128;
  float* out_tgt   = out_ref   + (size_t)Bb * Nn;
  float* out_cls   = out_tgt   + (size_t)Bb * Cc;
  float* out_off   = out_cls   + (size_t)Bb * Nn * Cc;

  char* ws = (char*)d_ws;
  half_t* H[4];
  for (int l = 0; l < 4; ++l) H[l] = (half_t*)(ws + (size_t)l * SZH);
  float*  SQ   = (float*)(ws + SQ_OFF);
  int*    IDXp = (int*)  (ws + IDX_OFF);
  half_t* PG[3];
  for (int l = 0; l < 3; ++l) PG[l] = (half_t*)(ws + PG_OFF + (size_t)l * PG_SZ);
  half_t* PF  = (half_t*)(ws + PF_OFF);
  half_t* PR0 = (half_t*)(ws + PR0_OFF);
  half_t* PR1 = (half_t*)(ws + PR1_OFF);
  half_t* PR2 = (half_t*)(ws + PR2_OFF);
  half_t* PC0 = (half_t*)(ws + PC0_OFF);
  half_t* PC1 = (half_t*)(ws + PC1_OFF);
  half_t* PC2 = (half_t*)(ws + PC2_OFF);
  half_t* PO0 = (half_t*)(ws + PO0_OFF);

  auto packB = [&](const float* W, half_t* P, int K, int N, int Kp, int Np) {
    const int threads = (Kp / 32) * (Np / 16) * 32;
    pack_b_kernel<<<(threads + 255) / 256, 256, 0, stream>>>(W, P, K, N, Kp, Np);
  };

  // -- pack all WMMA B operands once --
  for (int l = 0; l < 3; ++l) packB(gW[l], PG[l], 256, 128, 256, 128);
  packB(fW,  PF,  448, 128, 448, 128);
  packB(rW0, PR0, 128, 64, 128, 64);
  packB(rW1, PR1, 64,  64, 64,  64);
  packB(rW2, PR2, 64,  32, 64,  32);
  packB(cW0, PC0, 128, 64, 128, 64);
  packB(cW1, PC1, 64,  16, 64,  16);
  packB(cW2, PC2, 16,  Cc, 32,  32);   // zero-padded
  packB(oW0, PO0, 128, 64, 128, 64);

  // -- embed --
  embed_kernel<<<(Bb * Nn + 255) / 256, 256, 0, stream>>>(
      cls, pos, desc, cemb, pW0, pb0, pW1, pb1, pW2, pb2, pW3, pb3, H[0], SQ);

  // -- 3 x (kNN + EdgeConv) --
  for (int l = 0; l < Ll; ++l) {
    knn_kernel<<<Bb * (Nn / 16), 32, 0, stream>>>(H[l], SQ, IDXp);
    edgeconv_kernel<<<Bb * Nn / PPW, 32, 0, stream>>>(H[l], IDXp, PG[l], gB[l], H[l + 1], SQ);
  }

  // -- feat_mlp (-> d_out feats) --
  featmlp_kernel<<<Bb * Nn / 16, 32, 0, stream>>>(H[1], H[2], H[3], desc, PF, fB, out_feats);

  // -- heads --
  heads_kernel<<<Bb * Nn / 16, 32, 0, stream>>>(
      out_feats, PR0, rb0, PR1, rb1, PR2, rb2, rW3, rb3,
      PC0, cb0, PC1, cb1, PC2, cb2, PO0, ob0, oW1, ob1,
      out_ref, out_cls, out_off);
  tgt_kernel<<<1, 16, 0, stream>>>(desc, tW0, tb0, tW1, tb1, out_tgt);
}
